// GCN_69191923139115
// MI455X (gfx1250) — matrix operand order repaired and checked by reading, not compiled, not run on hardware
//
#include <hip/hip_runtime.h>
#include <hip/hip_bf16.h>

typedef __attribute__((ext_vector_type(16))) __bf16 v16bf;
typedef __attribute__((ext_vector_type(8)))  float  v8f;

#define IN_F  256
#define OUT_F 256
#define KSTEP 32
#define TILE_M 64            // rows per workgroup (4 row-strips x 16)
#define LDS_PAD 36           // 32 + 4 bf16 pad -> 72B row stride, conflict-free reads

__device__ __forceinline__ __bf16 f2bf(float f) {
    union { float f; unsigned u; } in; in.f = f;
    unsigned r = in.u + 0x7FFFu + ((in.u >> 16) & 1u);   // round-to-nearest-even
    unsigned short h = (unsigned short)(r >> 16);
    union { unsigned short u; __bf16 b; } out; out.u = h;
    return out.b;
}

// Guaranteed-native fp32 global atomic add (ISA §15.18 GLOBAL_ATOMIC_ADD_F32).
__device__ __forceinline__ void gatomic_add_f32(float* p, float v) {
    asm volatile("global_atomic_add_f32 %0, %1, off"
                 :
                 : "v"(p), "v"(v)
                 : "memory");
}

// support[N x 256] = X[N x 256] @ W[256 x 256], bf16 inputs / f32 accumulate via WMMA.
__global__ __launch_bounds__(256) void gcn_gemm_bf16(
    const float* __restrict__ X, const float* __restrict__ W,
    float* __restrict__ support, int nrows)
{
    __shared__ __bf16 WtS[OUT_F][LDS_PAD];   // transposed k-slice of W: WtS[n][kk]

    const int tid   = threadIdx.x;
    const int lane  = tid & 31;
    const int wave  = tid >> 5;              // 0..7
    const int wrow  = wave >> 1;             // 0..3  (16-row strip)
    const int wcol  = wave & 1;              // 0..1  (128-col half)
    const int mBase = blockIdx.x * TILE_M + wrow * 16;
    const int cBase = wcol * 128;
    const int l16   = lane & 15;

    v8f acc[8];
    #pragma unroll
    for (int j = 0; j < 8; ++j) acc[j] = (v8f){};

    // A-fragment row for this lane (clamped; out-of-range rows never stored)
    int arow = mBase + l16;
    if (arow >= nrows) arow = nrows - 1;
    const float* xr = X + (size_t)arow * IN_F;
    const int b1 = (lane < 16) ? 0 : 8;      // K offsets per 16-bit A layout
    const int b2 = b1 + 16;

    for (int k0 = 0; k0 < IN_F; k0 += KSTEP) {
        __syncthreads();
        // Stage W[k0:k0+32, 0:256] transposed into LDS as bf16.
        // iteration i: thread tid reads W[(k0+i)*256 + tid] (coalesced), writes WtS[tid][i].
        #pragma unroll 4
        for (int i = 0; i < KSTEP; ++i)
            WtS[tid][i] = f2bf(W[(size_t)(k0 + i) * OUT_F + tid]);
        __syncthreads();

        // Prefetch next k-slice (global_prefetch_b8) to overlap with the WMMA chain.
        if (k0 + KSTEP < IN_F) {
            __builtin_prefetch(&W[(size_t)(k0 + KSTEP) * OUT_F + tid], 0, 1);
            __builtin_prefetch(xr + k0 + KSTEP, 0, 1);
        }

        // A fragment: lane holds row M=l16; K = k0 + {b1..b1+7, b2..b2+7}
        const float4 f0 = *(const float4*)(xr + k0 + b1);
        const float4 f1 = *(const float4*)(xr + k0 + b1 + 4);
        const float4 f2 = *(const float4*)(xr + k0 + b2);
        const float4 f3 = *(const float4*)(xr + k0 + b2 + 4);
        v16bf a;
        a[0]=f2bf(f0.x); a[1]=f2bf(f0.y); a[2]=f2bf(f0.z); a[3]=f2bf(f0.w);
        a[4]=f2bf(f1.x); a[5]=f2bf(f1.y); a[6]=f2bf(f1.z); a[7]=f2bf(f1.w);
        a[8]=f2bf(f2.x); a[9]=f2bf(f2.y); a[10]=f2bf(f2.z); a[11]=f2bf(f2.w);
        a[12]=f2bf(f3.x); a[13]=f2bf(f3.y); a[14]=f2bf(f3.z); a[15]=f2bf(f3.w);

        const int kb = (lane < 16) ? 0 : 8;  // K base per 16-bit B layout
        #pragma unroll
        for (int j = 0; j < 8; ++j) {
            const int n = cBase + j * 16 + l16;   // B column for this lane
            v16bf b;
            #pragma unroll
            for (int i = 0; i < 16; ++i) b[i] = WtS[n][kb + i + (i & 8)];
            acc[j] = __builtin_amdgcn_wmma_f32_16x16x32_bf16(
                false, a, false, b, (short)0, acc[j], false, false);
        }
    }

    // D layout: lane<16 -> rows mBase+0..7, lane>=16 -> rows mBase+8..15; n = l16
    const int m0 = mBase + ((lane < 16) ? 0 : 8);
    if (mBase + 16 <= nrows) {
        // Fast path (uniform): full 16-row strip, unguarded stores.
        #pragma unroll
        for (int j = 0; j < 8; ++j) {
            float* colp = support + (size_t)m0 * OUT_F + cBase + j * 16 + l16;
            #pragma unroll
            for (int r = 0; r < 8; ++r)
                colp[(size_t)r * OUT_F] = acc[j][r];
        }
    } else {
        // Tail path: only the last (partial) block takes this.
        #pragma unroll
        for (int j = 0; j < 8; ++j) {
            const int n = cBase + j * 16 + l16;
            #pragma unroll
            for (int r = 0; r < 8; ++r) {
                const int m = m0 + r;
                if (m < nrows) support[(size_t)m * OUT_F + n] = acc[j][r];
            }
        }
    }
}

// out[dst[e]] += vals[e] * support[src[e]]   (wave per edge, 8 floats/lane)
__global__ __launch_bounds__(256) void gcn_scatter(
    const float* __restrict__ support, const float* __restrict__ vals,
    const int* __restrict__ src, const int* __restrict__ dst,
    float* __restrict__ out, int E)
{
    const int wave = blockIdx.x * (blockDim.x >> 5) + (threadIdx.x >> 5);
    const int lane = threadIdx.x & 31;
    if (wave >= E) return;

    const int   s = src[wave];
    const int   d = dst[wave];
    const float v = vals[wave];

    const float4* srow = (const float4*)(support + (size_t)s * OUT_F);
    float*        drow = out + (size_t)d * OUT_F;

    const float4 p0 = srow[lane * 2 + 0];
    const float4 p1 = srow[lane * 2 + 1];
    float* dp = drow + lane * 8;
    gatomic_add_f32(dp + 0, v * p0.x);
    gatomic_add_f32(dp + 1, v * p0.y);
    gatomic_add_f32(dp + 2, v * p0.z);
    gatomic_add_f32(dp + 3, v * p0.w);
    gatomic_add_f32(dp + 4, v * p1.x);
    gatomic_add_f32(dp + 5, v * p1.y);
    gatomic_add_f32(dp + 6, v * p1.z);
    gatomic_add_f32(dp + 7, v * p1.w);
}

extern "C" void kernel_launch(void* const* d_in, const int* in_sizes, int n_in,
                              void* d_out, int out_size, void* d_ws, size_t ws_size,
                              hipStream_t stream) {
    const float* x  = (const float*)d_in[0];
    const float* W  = (const float*)d_in[1];
    const float* ev = (const float*)d_in[2];
    const int*   es = (const int*)d_in[3];
    const int*   ed = (const int*)d_in[4];

    const int nrows = in_sizes[0] / IN_F;   // 100000
    const int E     = in_sizes[2];          // 3200000

    float* support = (float*)d_ws;          // needs nrows*256*4 = 102.4 MB

    const int gridM = (nrows + TILE_M - 1) / TILE_M;
    gcn_gemm_bf16<<<gridM, 256, 0, stream>>>(x, W, support, nrows);

    hipMemsetAsync(d_out, 0, (size_t)out_size * sizeof(float), stream);

    const int edgesPerBlock = 256 / 32;     // 8 waves
    const int gridE = (E + edgesPerBlock - 1) / edgesPerBlock;
    gcn_scatter<<<gridE, 256, 0, stream>>>(support, ev, es, ed, (float*)d_out, E);
}
